// CPN_67740224192953
// MI455X (gfx1250) — matrix operand order repaired and checked by reading, not compiled, not run on hardware
//
#include <hip/hip_runtime.h>
#include <hip/hip_bf16.h>

typedef __attribute__((ext_vector_type(16))) _Float16 v16h;
typedef __attribute__((ext_vector_type(8)))  _Float16 v8h;
typedef __attribute__((ext_vector_type(8)))  float    v8f;
typedef int v4i_ __attribute__((vector_size(16)));
typedef __attribute__((address_space(1))) v4i_* gvec_ptr;
typedef __attribute__((address_space(3))) v4i_* lvec_ptr;

#define B_    4
#define CIN_  3
#define H_    512
#define W_    512
#define C_    64
#define NDET_ 512
#define SMP_  32
#define HP_   (H_ + 6)
#define WP_   (W_ + 6)
#define HW_   (H_ * W_)
#define NCH_  26          // 2 score + 2 loc + 20 fourier + 2 refine
#define NPAD_ 32

// ---------------------------------------------------------------------------
// 0) zero-fill (padded feature borders must be zero)
// ---------------------------------------------------------------------------
__global__ void fill_zero_kernel(uint4* __restrict__ p, long n) {
  long i = (long)blockIdx.x * blockDim.x + threadIdx.x;
  if (i < n) { uint4 z; z.x = 0; z.y = 0; z.z = 0; z.w = 0; p[i] = z; }
}

__global__ void zero_cnt_kernel(int* __restrict__ cnt) {
  int t = threadIdx.x;
  if (t < B_) cnt[t] = 0;
}

// ---------------------------------------------------------------------------
// 1) backbone: 3x3 conv 3->64 + relu, write fp16 NHWC padded (pad=3)
// ---------------------------------------------------------------------------
__global__ __launch_bounds__(256) void backbone_kernel(
    const float* __restrict__ x, const float* __restrict__ w_bb,
    const float* __restrict__ b_bb, _Float16* __restrict__ fpad) {
  __shared__ float swbb[1728];            // 64*27
  int tid = threadIdx.x;
  for (int i = tid; i < 1728; i += 256) swbb[i] = w_bb[i];
  __syncthreads();

  int pix = blockIdx.x * 64 + (tid >> 2);
  int cg  = tid & 3;                      // 16-channel group
  int b   = pix >> 18;                    // / (512*512)
  int rem = pix & (HW_ - 1);
  int y   = rem >> 9;
  int xx  = rem & 511;

  float in[27];
#pragma unroll
  for (int ci = 0; ci < 3; ++ci)
#pragma unroll
    for (int ky = 0; ky < 3; ++ky)
#pragma unroll
      for (int kx = 0; kx < 3; ++kx) {
        int iy = y + ky - 1, ix = xx + kx - 1;
        float v = 0.0f;
        if (iy >= 0 && iy < H_ && ix >= 0 && ix < W_)
          v = x[(((long)b * 3 + ci) * H_ + iy) * W_ + ix];
        in[ci * 9 + ky * 3 + kx] = v;
      }

  long obase = (((long)b * HP_ + (y + 3)) * WP_ + (xx + 3)) * C_ + cg * 16;
#pragma unroll
  for (int co = 0; co < 16; ++co) {
    int oc = cg * 16 + co;
    float acc = b_bb[oc];
    const float* wp = swbb + oc * 27;
#pragma unroll
    for (int j = 0; j < 27; ++j) acc += wp[j] * in[j];
    fpad[obase + co] = (_Float16)fmaxf(acc, 0.0f);
  }
}

// ---------------------------------------------------------------------------
// 2) repack head weights fp32 OIHW -> fp16 B-matrix layout
//    wr[((tap*2+s)*32 + n)*32 + k],  channel = s*32+k,  tap = dy*7+dx
// ---------------------------------------------------------------------------
__global__ void repack_kernel(const float* __restrict__ ws_, const float* __restrict__ wl,
                              const float* __restrict__ wf, const float* __restrict__ wrf,
                              _Float16* __restrict__ wr) {
  int t = blockIdx.x * 256 + threadIdx.x;       // 49*2048 = 100352 exactly
  int k   = t & 31;
  int n   = (t >> 5) & 31;
  int s   = (t >> 10) & 1;
  int tap = t >> 11;
  int ch  = s * 32 + k;
  float v = 0.0f;
  if (n < 2)        v = ws_[((long)n * C_ + ch) * 49 + tap];
  else if (n < 4)   v = wl[((long)(n - 2) * C_ + ch) * 49 + tap];
  else if (n < 24)  v = wf[((long)(n - 4) * C_ + ch) * 49 + tap];
  else if (n < 26)  v = wrf[((long)(n - 24) * C_ + ch) * 49 + tap];
  wr[t] = (_Float16)v;
}

// ---------------------------------------------------------------------------
// 3) heads: implicit-GEMM WMMA. 256 threads = 8 waves, 1 16-pixel tile/wave.
//    Per tile: 49 taps x 2 k-steps x 2 n-tiles = 196 v_wmma_f32_16x16x32_f16
//    Weight staging: async global->LDS copies tracked by ASYNCcnt.
// ---------------------------------------------------------------------------
__global__ __launch_bounds__(256) void heads_kernel(
    const _Float16* __restrict__ fpad, const _Float16* __restrict__ wr,
    const float* __restrict__ bs, const float* __restrict__ bl,
    const float* __restrict__ bf, const float* __restrict__ br,
    float* __restrict__ oh) {
  __shared__ _Float16 sw[7 * 2048];    // one tap-row of repacked weights (28KB)
  __shared__ float sbias[NPAD_];

  int tid = threadIdx.x;
  if (tid < NPAD_) {
    float v = 0.0f;
    if (tid < 2)       v = bs[tid];
    else if (tid < 4)  v = bl[tid - 2];
    else if (tid < 24) v = bf[tid - 4];
    else if (tid < 26) v = br[tid - 24];
    sbias[tid] = v;
  }

  int wid  = tid >> 5;
  int lane = tid & 31;
  int tile = blockIdx.x * 8 + wid;      // 65536 tiles total
  int b    = tile >> 14;                // / (512*32)
  int rem  = tile & 16383;
  int y    = rem >> 5;
  int x0   = (rem & 31) << 4;

  v8f c0 = {}, c1 = {};
  int m     = lane & 15;
  int khalf = (lane >> 4) << 4;         // 0 or 16 (A-fragment contiguous k half)
  const long rowstride = (long)WP_ * C_;
  const _Float16* fbase = fpad + ((long)b * HP_ + y) * rowstride
                               + (long)(x0 + m) * C_ + khalf;

  for (int dy = 0; dy < 7; ++dy) {
    __syncthreads();
    {   // stage 7 taps * 2048 halves of weights into LDS
      const _Float16* src = wr + (long)dy * 7 * 2048;
#if __has_builtin(__builtin_amdgcn_global_load_async_to_lds_b128)
#pragma unroll
      for (int i = 0; i < 7; ++i) {
        __builtin_amdgcn_global_load_async_to_lds_b128(
            (gvec_ptr)(src + tid * 8 + i * 2048),
            (lvec_ptr)(sw + tid * 8 + i * 2048),
            0, 0);
      }
#if __has_builtin(__builtin_amdgcn_s_wait_asynccnt)
      __builtin_amdgcn_s_wait_asynccnt(0);
#else
      asm volatile("s_wait_asynccnt 0" ::: "memory");
#endif
#else
      const uint32_t* s32 = (const uint32_t*)src;
      uint32_t* dst = (uint32_t*)sw;
      for (int i = tid; i < 7 * 1024; i += 256) dst[i] = s32[i];
#endif
    }
    __syncthreads();

    const _Float16* arow = fbase + (long)dy * rowstride;
    if (dy < 6) __builtin_prefetch((const void*)(arow + rowstride), 0, 1);

#pragma unroll
    for (int dx = 0; dx < 7; ++dx) {
      const _Float16* ap = arow + dx * C_;
#pragma unroll
      for (int s = 0; s < 2; ++s) {
        v8h lo = *(const v8h*)(ap + 32 * s);
        v8h hi = *(const v8h*)(ap + 32 * s + 8);
        v16h a = __builtin_shufflevector(lo, hi, 0, 1, 2, 3, 4, 5, 6, 7,
                                         8, 9, 10, 11, 12, 13, 14, 15);
        const _Float16* bp = &sw[(((dx * 2 + s) * 32) + (lane & 15)) * 32 + (lane >> 4) * 16];
        v16h b0 = *(const v16h*)(bp);
        v16h b1 = *(const v16h*)(bp + 512);
        c0 = __builtin_amdgcn_wmma_f32_16x16x32_f16(false, a, false, b0,
                                                    (short)0, c0, false, false);
        c1 = __builtin_amdgcn_wmma_f32_16x16x32_f16(false, a, false, b1,
                                                    (short)0, c1, false, false);
      }
    }
  }

  // epilogue: bias + ScaledTanh on refinement channels, scatter to NCHW f32
  long obase = (long)b * NCH_ * HW_ + (long)y * W_ + x0;
  int nn = lane & 15;
  int mb = (lane >> 4) << 3;
#pragma unroll
  for (int r = 0; r < 8; ++r) {
    int mm = r + mb;
    float v0 = c0[r] + sbias[nn];
    float v1 = c1[r] + sbias[16 + nn];
    oh[obase + (long)nn * HW_ + mm] = v0;            // channels 0..15
    int n1 = 16 + nn;
    if (n1 < 24)       oh[obase + (long)n1 * HW_ + mm] = v1;                 // fourier
    else if (n1 < 26)  oh[obase + (long)n1 * HW_ + mm] = 3.0f * tanhf(v1);   // refine
  }
}

// ---------------------------------------------------------------------------
// 4) per-image top-k threshold by binary search on (s1-s0)
// ---------------------------------------------------------------------------
__global__ __launch_bounds__(256) void thresh_kernel(const float* __restrict__ oh,
                                                     float* __restrict__ thr) {
  int b = blockIdx.x, tid = threadIdx.x;
  const float* s0 = oh + (long)b * NCH_ * HW_;
  const float* s1 = s0 + HW_;
  __shared__ float rf[256];
  __shared__ int   ri[256];
  __shared__ float sb[2];

  float lmin = 3.4e38f, lmax = -3.4e38f;
  for (int i = tid; i < HW_; i += 256) {
    float d = s1[i] - s0[i];
    lmin = fminf(lmin, d); lmax = fmaxf(lmax, d);
  }
  rf[tid] = lmax; __syncthreads();
  for (int s = 128; s > 0; s >>= 1) { if (tid < s) rf[tid] = fmaxf(rf[tid], rf[tid + s]); __syncthreads(); }
  if (tid == 0) sb[1] = rf[0];
  __syncthreads();
  rf[tid] = lmin; __syncthreads();
  for (int s = 128; s > 0; s >>= 1) { if (tid < s) rf[tid] = fminf(rf[tid], rf[tid + s]); __syncthreads(); }
  if (tid == 0) sb[0] = rf[0];
  __syncthreads();

  float lo = sb[0], hi = sb[1];
  for (int it = 0; it < 20; ++it) {
    float mid = 0.5f * (lo + hi);
    int c = 0;
    for (int i = tid; i < HW_; i += 256) c += (s1[i] - s0[i]) > mid;
    ri[tid] = c; __syncthreads();
    for (int s = 128; s > 0; s >>= 1) { if (tid < s) ri[tid] += ri[tid + s]; __syncthreads(); }
    if (tid == 0) { if (ri[0] > NDET_) sb[0] = mid; else sb[1] = mid; }
    __syncthreads();
    lo = sb[0]; hi = sb[1];
  }
  if (tid == 0) thr[b] = lo;   // count(d > lo) >= NDET_; compaction clamps
}

// ---------------------------------------------------------------------------
// 5) compact detections above threshold, 6) fill any shortage
// ---------------------------------------------------------------------------
__global__ void compact_kernel(const float* __restrict__ oh, const float* __restrict__ thr,
                               int* __restrict__ cnt, int* __restrict__ idx) {
  int b = blockIdx.x >> 10;                       // 1024 blocks per image
  int p = ((blockIdx.x & 1023) << 8) | threadIdx.x;
  const float* s0 = oh + (long)b * NCH_ * HW_;
  float d = s0[HW_ + p] - s0[p];
  if (d > thr[b]) {
    int slot = atomicAdd(cnt + b, 1);
    if (slot < NDET_) idx[(b << 9) + slot] = p;
  }
}

__global__ void fill_idx_kernel(int* __restrict__ idx, const int* __restrict__ cnt) {
  int b = blockIdx.x, i = threadIdx.x;
  int c = cnt[b]; if (c > NDET_) c = NDET_;
  if (i >= c) idx[(b << 9) + i] = i;
}

// ---------------------------------------------------------------------------
// 7) fourier contour synthesis + 4 gather-refine iterations
// ---------------------------------------------------------------------------
__global__ __launch_bounds__(256) void contour_kernel(const float* __restrict__ oh,
                                                      const int* __restrict__ idx,
                                                      float* __restrict__ out) {
  int t = blockIdx.x * 256 + threadIdx.x;         // B*NDET*SMP = 65536
  int sIdx = t & 31;
  int n    = (t >> 5) & 511;
  int b    = t >> 14;
  int p = idx[(b << 9) + n];
  const float* base = oh + (long)b * NCH_ * HW_;
  float px = (float)(p & 511);
  float py = (float)(p >> 9);
  float xs = px + base[2 * HW_ + p];              // cx
  float ys = py + base[3 * HW_ + p];              // cy
  float tt = (float)sIdx * (1.0f / 32.0f);
#pragma unroll
  for (int o = 1; o <= 5; ++o) {
    float ang = 6.28318530718f * (float)o * tt;
    float sa, ca;
    __sincosf(ang, &sa, &ca);
    const float* cf = base + (long)(4 + (o - 1) * 4) * HW_ + p;
    xs += cf[0] * ca + cf[HW_] * sa;
    ys += cf[2 * (long)HW_] * ca + cf[3 * (long)HW_] * sa;
  }
  const float* rx = base + 24 * (long)HW_;
  const float* ry = base + 25 * (long)HW_;
  float xc = xs, yc = ys;
#pragma unroll
  for (int it = 0; it < 4; ++it) {
    float xi = fminf(fmaxf(roundf(xc), 0.0f), 511.0f);
    float yi = fminf(fmaxf(roundf(yc), 0.0f), 511.0f);
    int lin = (int)yi * W_ + (int)xi;
    xc = xi + rx[lin];
    yc = yi + ry[lin];
  }
  long o2 = (long)t * 2;
  out[o2] = xc; out[o2 + 1] = yc;
}

// ---------------------------------------------------------------------------
// launcher
// ---------------------------------------------------------------------------
extern "C" void kernel_launch(void* const* d_in, const int* in_sizes, int n_in,
                              void* d_out, int out_size, void* d_ws, size_t ws_size,
                              hipStream_t stream) {
  const float* x        = (const float*)d_in[0];
  const float* w_bb     = (const float*)d_in[1];
  const float* b_bb     = (const float*)d_in[2];
  const float* w_score  = (const float*)d_in[3];
  const float* b_score  = (const float*)d_in[4];
  const float* w_loc    = (const float*)d_in[5];
  const float* b_loc    = (const float*)d_in[6];
  const float* w_four   = (const float*)d_in[7];
  const float* b_four   = (const float*)d_in[8];
  const float* w_ref    = (const float*)d_in[9];
  const float* b_ref    = (const float*)d_in[10];
  (void)in_sizes; (void)n_in; (void)out_size; (void)ws_size;

  char* ws = (char*)d_ws;
  size_t off = 0;
  auto take = [&](size_t bytes) { size_t o = off; off = (off + bytes + 255) & ~(size_t)255; return o; };

  const long FPAD_BYTES = (long)B_ * HP_ * WP_ * C_ * 2;   // 137,381,888
  const long OH_BYTES   = (long)B_ * NCH_ * HW_ * 4;       // 109,051,904
  const long WR_BYTES   = 49L * 64 * 32 * 2;               // 200,704

  _Float16* fpad = (_Float16*)(ws + take(FPAD_BYTES));
  _Float16* wrep = (_Float16*)(ws + take(WR_BYTES));
  float*    oh   = (float*)(ws + take(OH_BYTES));
  float*    thr  = (float*)(ws + take(B_ * sizeof(float)));
  int*      cnt  = (int*)(ws + take(B_ * sizeof(int)));
  int*      idx  = (int*)(ws + take((size_t)B_ * NDET_ * sizeof(int)));

  // 0) zero padded features + counters
  long n16 = FPAD_BYTES / 16;
  fill_zero_kernel<<<(int)((n16 + 255) / 256), 256, 0, stream>>>((uint4*)fpad, n16);
  zero_cnt_kernel<<<1, 32, 0, stream>>>(cnt);

  // 1) backbone -> fp16 NHWC padded
  backbone_kernel<<<(B_ * HW_) / 64, 256, 0, stream>>>(x, w_bb, b_bb, fpad);

  // 2) repack head weights
  repack_kernel<<<392, 256, 0, stream>>>(w_score, w_loc, w_four, w_ref, wrep);

  // 3) WMMA implicit-GEMM heads
  heads_kernel<<<8192, 256, 0, stream>>>(fpad, wrep, b_score, b_loc, b_four, b_ref, oh);

  // 4-6) top-k selection
  thresh_kernel<<<B_, 256, 0, stream>>>(oh, thr);
  compact_kernel<<<B_ * (HW_ / 256), 256, 0, stream>>>(oh, thr, cnt, idx);
  fill_idx_kernel<<<B_, NDET_, 0, stream>>>(idx, cnt);

  // 7) contour synthesis + refinement
  contour_kernel<<<(B_ * NDET_ * SMP_) / 256, 256, 0, stream>>>(oh, idx, (float*)d_out);
}
